// PSA_66219805770224
// MI455X (gfx1250) — compile-verified, matrix operand order checked by reading, not run on hardware
//
#include <hip/hip_runtime.h>
#include <hip/hip_bf16.h>
#include <math.h>

// ---------------------------------------------------------------------------
// CDNA5 (gfx1250) PSA pipeline: pre-converted bf16 operands + WMMA GEMMs,
// register-double-buffered global->LDS staging.
// ---------------------------------------------------------------------------

typedef __attribute__((ext_vector_type(16))) __bf16        v16bf;
typedef __attribute__((ext_vector_type(8)))  float         v8f;
typedef __attribute__((ext_vector_type(8)))  unsigned int  v8u;

#define B_   4
#define N_   16384     // 128*128
#define C_   256
#define H_   8
#define D_   32

__device__ __forceinline__ unsigned short f2bf(float f) {
    unsigned int u = __float_as_uint(f);
    unsigned int r = u + 0x7FFFu + ((u >> 16) & 1u);   // round-to-nearest-even
    return (unsigned short)(r >> 16);
}

__device__ __forceinline__ unsigned int pack_bf2(float a, float b) {
    return (unsigned int)f2bf(a) | ((unsigned int)f2bf(b) << 16);
}

__device__ __forceinline__ v8f zero8() {
    v8f z = {0.f,0.f,0.f,0.f,0.f,0.f,0.f,0.f};
    return z;
}

__device__ __forceinline__ v8f wmma_bf16(v16bf a, v16bf b, v8f c) {
    return __builtin_amdgcn_wmma_f32_16x16x32_bf16(
        false, a, false, b, (short)0, c, false, false);
}

// 16x32 bf16 A/B fragment from a row-major LDS tile (row = M or N, cols = K).
// lane L -> row = L%16; chunk i holds K = ((L>=16)?8:0)+((i>=4)?16:0)+2*(i&3).
__device__ __forceinline__ v16bf load_frag16(const unsigned short* base,
                                             int stride_elems, int lane) {
    const unsigned short* rp = base + (size_t)(lane & 15) * stride_elems;
    int khalf = (lane >> 4) << 3;
    v8u u;
#pragma unroll
    for (int i = 0; i < 8; ++i) {
        int kb = khalf + ((i >= 4) ? 16 : 0) + ((i & 3) << 1);
        u[i] = *(const unsigned int*)(rp + kb);
    }
    return __builtin_bit_cast(v16bf, u);
}

// ---------------------------------------------------------------------------
// Kernel 0: bulk fp32 -> bf16 conversion (vectorized, bandwidth-bound).
// ---------------------------------------------------------------------------
__global__ __launch_bounds__(256)
void cvt_bf16_kernel(const float* __restrict__ s, unsigned short* __restrict__ d,
                     int n4) {
    int i = blockIdx.x * 256 + threadIdx.x;
    if (i < n4) {
        float4 v = ((const float4*)s)[i];
        uint2 o;
        o.x = pack_bf2(v.x, v.y);
        o.y = pack_bf2(v.z, v.w);
        ((uint2*)d)[i] = o;
    }
}

// ---------------------------------------------------------------------------
// Kernel 1: Out[M,256] f32 = A_bf16[M,256] @ W_bf16[256,256]^T (+bias).
// 256 threads = 8 waves, tile 128(M) x 128(N), K-step 32. Wave grid 4x2:
// each wave owns 32x64 = 2x4 fragments -> 8 WMMA per K-step.
// Register double-buffering hides global load latency behind the WMMAs.
// ---------------------------------------------------------------------------
__global__ __launch_bounds__(256)
void gemm256_kernel(const unsigned short* __restrict__ A,
                    const unsigned short* __restrict__ W,
                    const float* __restrict__ bias, float* __restrict__ Out,
                    int M) {
    const int mBase = blockIdx.x * 128;
    const int nBase = blockIdx.y * 128;
    const int tid  = threadIdx.x;
    const int lane = tid & 31;
    const int wave = tid >> 5;
    const int wm = wave >> 1;      // M offset 32*wm
    const int wn = wave & 1;       // N offset 64*wn

    __shared__ __align__(16) unsigned short As[128][36];  // bf16, 72B rows
    __shared__ __align__(16) unsigned short Bs[128][36];

    v8f acc[2][4];
#pragma unroll
    for (int mi = 0; mi < 2; ++mi)
#pragma unroll
        for (int ni = 0; ni < 4; ++ni) acc[mi][ni] = zero8();

    const int r0  = tid >> 3;          // 0..31
    const int cg4 = (tid & 7) << 2;    // 0,4,...,28 (4 bf16 = 8 bytes)

    uint2 ra[4], rb[4];
#pragma unroll
    for (int t = 0; t < 4; ++t) {      // fixed trip count: no exec masking
        int rr = r0 + 32 * t;
        ra[t] = *(const uint2*)&A[(size_t)(mBase + rr) * C_ + cg4];
        rb[t] = *(const uint2*)&W[(size_t)(nBase + rr) * C_ + cg4];
    }

#pragma unroll 1
    for (int kk = 0; kk < 256; kk += 32) {
#pragma unroll
        for (int t = 0; t < 4; ++t) {
            int rr = r0 + 32 * t;
            *(uint2*)&As[rr][cg4] = ra[t];
            *(uint2*)&Bs[rr][cg4] = rb[t];
        }
        __syncthreads();

        if (kk + 32 < 256) {           // issue next tile loads now; the wait
#pragma unroll                         // lands at next iteration's ds_store
            for (int t = 0; t < 4; ++t) {
                int rr = r0 + 32 * t;
                ra[t] = *(const uint2*)&A[(size_t)(mBase + rr) * C_ + kk + 32 + cg4];
                rb[t] = *(const uint2*)&W[(size_t)(nBase + rr) * C_ + kk + 32 + cg4];
            }
            if (kk + 64 < 256) {       // prefetch one more tile ahead
                __builtin_prefetch(&A[(size_t)(mBase + r0) * C_ + kk + 64 + cg4], 0, 3);
                __builtin_prefetch(&W[(size_t)(nBase + r0) * C_ + kk + 64 + cg4], 0, 3);
            }
        }

        v16bf a0 = load_frag16(&As[32 * wm +  0][0], 36, lane);
        v16bf a1 = load_frag16(&As[32 * wm + 16][0], 36, lane);
        v16bf b0 = load_frag16(&Bs[64 * wn +  0][0], 36, lane);
        v16bf b1 = load_frag16(&Bs[64 * wn + 16][0], 36, lane);
        v16bf b2 = load_frag16(&Bs[64 * wn + 32][0], 36, lane);
        v16bf b3 = load_frag16(&Bs[64 * wn + 48][0], 36, lane);
        acc[0][0] = wmma_bf16(a0, b0, acc[0][0]);
        acc[0][1] = wmma_bf16(a0, b1, acc[0][1]);
        acc[0][2] = wmma_bf16(a0, b2, acc[0][2]);
        acc[0][3] = wmma_bf16(a0, b3, acc[0][3]);
        acc[1][0] = wmma_bf16(a1, b0, acc[1][0]);
        acc[1][1] = wmma_bf16(a1, b1, acc[1][1]);
        acc[1][2] = wmma_bf16(a1, b2, acc[1][2]);
        acc[1][3] = wmma_bf16(a1, b3, acc[1][3]);
        __syncthreads();
    }

    const int col0 = lane & 15;
    const int rb2  = (lane >> 4) << 3;
#pragma unroll
    for (int mi = 0; mi < 2; ++mi) {
#pragma unroll
        for (int ni = 0; ni < 4; ++ni) {
            int col = nBase + 64 * wn + 16 * ni + col0;
            float bv = bias ? bias[col] : 0.f;
#pragma unroll
            for (int r = 0; r < 8; ++r) {
                int row = mBase + 32 * wm + 16 * mi + rb2 + r;
                Out[(size_t)row * C_ + col] = acc[mi][ni][r] + bv;
            }
        }
    }
}

// ---------------------------------------------------------------------------
// Kernel 2: L2-normalize along n and transpose [b,n,c] f32 -> [b*c, n] bf16.
// ---------------------------------------------------------------------------
__global__ __launch_bounds__(256)
void l2norm_kernel(const float* __restrict__ P, unsigned short* __restrict__ T) {
    const int bc = blockIdx.x;
    const int b  = bc >> 8;
    const int ch = bc & 255;
    const float* base = P + (size_t)b * N_ * C_ + ch;
    const int tid = threadIdx.x;

    float ss = 0.f;
    for (int i = tid; i < N_; i += 256) {
        float v = base[(size_t)i * C_];
        ss += v * v;
    }
    __shared__ float red[256];
    red[tid] = ss;
    __syncthreads();
#pragma unroll
    for (int s = 128; s > 0; s >>= 1) {
        if (tid < s) red[tid] += red[tid + s];
        __syncthreads();
    }
    const float inv = 1.0f / fmaxf(sqrtf(red[0]), 1e-12f);

    unsigned short* out = T + (size_t)bc * N_;
    for (int i = tid; i < N_; i += 256)
        out[i] = f2bf(base[(size_t)i * C_] * inv);
}

// ---------------------------------------------------------------------------
// Kernel 3: per (b,h): S[d][e] = sum_n K[d,n]*Q[e,n] over n=16384 via WMMA
// (8 waves on disjoint K-stripes, register double-buffered), then scale +
// row softmax over e. Output attn[b,h,32,32] f32.
// ---------------------------------------------------------------------------
__global__ __launch_bounds__(256)
void qk_softmax_kernel(const unsigned short* __restrict__ Kn,
                       const unsigned short* __restrict__ Qn,
                       const float* __restrict__ rescale,
                       float* __restrict__ attnOut) {
    const int bh = blockIdx.x;
    const int b = bh >> 3, h = bh & 7;
    const unsigned short* Kt = Kn + (size_t)(b * C_ + h * D_) * N_;
    const unsigned short* Qt = Qn + (size_t)(b * C_ + h * D_) * N_;
    const int tid  = threadIdx.x;
    const int lane = tid & 31;
    const int wave = tid >> 5;

    __shared__ __align__(16) unsigned short Ksh[8][32][40];
    __shared__ __align__(16) unsigned short Qsh[8][32][40];
    __shared__ float Psum[8][32][32];
    __shared__ float Srow[32][33];

    v8f a00 = zero8(), a01 = zero8(), a10 = zero8(), a11 = zero8();

    uint4 rk[4], rq[4];
    {
        const int k0 = wave * 32;
        const uint4* gk = (const uint4*)(Kt + (size_t)lane * N_ + k0);
        const uint4* gq = (const uint4*)(Qt + (size_t)lane * N_ + k0);
#pragma unroll
        for (int j = 0; j < 4; ++j) { rk[j] = gk[j]; rq[j] = gq[j]; }
    }

#pragma unroll 1
    for (int it = 0; it < 64; ++it) {
        uint4* lk = (uint4*)&Ksh[wave][lane][0];
        uint4* lq = (uint4*)&Qsh[wave][lane][0];
#pragma unroll
        for (int j = 0; j < 4; ++j) { lk[j] = rk[j]; lq[j] = rq[j]; }

        if (it + 1 < 64) {             // next stripe loads overlap the WMMAs
            const int k0 = ((it + 1) * 8 + wave) * 32;
            const uint4* gk = (const uint4*)(Kt + (size_t)lane * N_ + k0);
            const uint4* gq = (const uint4*)(Qt + (size_t)lane * N_ + k0);
#pragma unroll
            for (int j = 0; j < 4; ++j) { rk[j] = gk[j]; rq[j] = gq[j]; }
        }

        v16bf ka0 = load_frag16(&Ksh[wave][ 0][0], 40, lane);
        v16bf ka1 = load_frag16(&Ksh[wave][16][0], 40, lane);
        v16bf qb0 = load_frag16(&Qsh[wave][ 0][0], 40, lane);
        v16bf qb1 = load_frag16(&Qsh[wave][16][0], 40, lane);
        a00 = wmma_bf16(ka0, qb0, a00);
        a01 = wmma_bf16(ka0, qb1, a01);
        a10 = wmma_bf16(ka1, qb0, a10);
        a11 = wmma_bf16(ka1, qb1, a11);
    }

    const int col0 = lane & 15;
    const int rb   = (lane >> 4) << 3;
    v8f accs[2][2] = {{a00, a01}, {a10, a11}};
#pragma unroll
    for (int mi = 0; mi < 2; ++mi)
#pragma unroll
        for (int ni = 0; ni < 2; ++ni)
#pragma unroll
            for (int r = 0; r < 8; ++r)
                Psum[wave][16 * mi + rb + r][16 * ni + col0] = accs[mi][ni][r];
    __syncthreads();

    const float rs = rescale[h];
#pragma unroll
    for (int e4 = 0; e4 < 4; ++e4) {
        int idx = tid * 4 + e4;
        int d = idx >> 5, e = idx & 31;
        float s = 0.f;
#pragma unroll
        for (int w = 0; w < 8; ++w) s += Psum[w][d][e];
        Srow[d][e] = s * rs;
    }
    __syncthreads();

    if (tid < 32) {
        int d = tid;
        float m = -3.0e38f;
#pragma unroll
        for (int e = 0; e < 32; ++e) m = fmaxf(m, Srow[d][e]);
        float ex[32];
        float sum = 0.f;
#pragma unroll
        for (int e = 0; e < 32; ++e) { ex[e] = expf(Srow[d][e] - m); sum += ex[e]; }
        float inv = 1.0f / sum;
        float* o = attnOut + (size_t)bh * 1024 + d * 32;
#pragma unroll
        for (int e = 0; e < 32; ++e) o[e] = ex[e] * inv;
    }
}

// ---------------------------------------------------------------------------
// Kernel 4: xo = attny @ attn (32x32x32 per (b,h)), store bf16.
// ---------------------------------------------------------------------------
__global__ __launch_bounds__(32)
void xo_kernel(const float* __restrict__ attnY, const float* __restrict__ attnX,
               unsigned short* __restrict__ xo2b) {
    const int bh = blockIdx.x;
    const int d  = threadIdx.x;
    __shared__ float X[32][33];
    for (int i = d; i < 1024; i += 32)
        X[i >> 5][i & 31] = attnX[(size_t)bh * 1024 + i];
    __syncthreads();
    float yrow[32];
#pragma unroll
    for (int e = 0; e < 32; ++e)
        yrow[e] = attnY[(size_t)bh * 1024 + d * 32 + e];
#pragma unroll
    for (int f = 0; f < 32; ++f) {
        float s = 0.f;
#pragma unroll
        for (int e = 0; e < 32; ++e) s += yrow[e] * X[e][f];
        xo2b[(size_t)bh * 1024 + d * 32 + f] = f2bf(s);
    }
}

// ---------------------------------------------------------------------------
// Kernel 5: out[b,n,h*32+d] = sum_f xo[b,h,d,f] * V[b,n,h*32+f]; bf16 output
// feeds the projection GEMM directly.
// ---------------------------------------------------------------------------
__global__ __launch_bounds__(256)
void av_kernel(const unsigned short* __restrict__ xo2b,
               const float* __restrict__ Pv, unsigned short* __restrict__ OutAb) {
    const int bh = blockIdx.y;
    const int b = bh >> 3, h = bh & 7;
    const int nBase = blockIdx.x * 128;
    const int tid  = threadIdx.x;
    const int lane = tid & 31;
    const int wave = tid >> 5;

    __shared__ __align__(16) unsigned short Axo[32][40];
    __shared__ __align__(16) unsigned short Vs[128][40];
    __shared__ float Ot[32][132];

#pragma unroll
    for (int j = 0; j < 4; ++j) {
        int idx = tid * 4 + j;
        Axo[idx >> 5][idx & 31] = xo2b[(size_t)bh * 1024 + idx];
    }
    {
        const int n0 = tid >> 3;
        const int fg = (tid & 7) << 2;
#pragma unroll
        for (int t = 0; t < 4; ++t) {
            int nn = n0 + 32 * t;
            float4 v = *(const float4*)
                &Pv[((size_t)(b * N_ + nBase + nn)) * C_ + h * D_ + fg];
            Vs[nn][fg + 0] = f2bf(v.x);
            Vs[nn][fg + 1] = f2bf(v.y);
            Vs[nn][fg + 2] = f2bf(v.z);
            Vs[nn][fg + 3] = f2bf(v.w);
        }
    }
    __syncthreads();

    v16bf bfr = load_frag16(&Vs[wave * 16][0], 40, lane);
    v16bf a0  = load_frag16(&Axo[ 0][0], 40, lane);
    v16bf a1  = load_frag16(&Axo[16][0], 40, lane);
    v8f c0 = wmma_bf16(a0, bfr, zero8());
    v8f c1 = wmma_bf16(a1, bfr, zero8());

    const int col0 = lane & 15;
    const int rb   = (lane >> 4) << 3;
#pragma unroll
    for (int r = 0; r < 8; ++r) {
        Ot[ 0 + rb + r][wave * 16 + col0] = c0[r];
        Ot[16 + rb + r][wave * 16 + col0] = c1[r];
    }
    __syncthreads();

    const int n0 = tid >> 3;
    const int dg = (tid & 7) << 2;
#pragma unroll
    for (int t = 0; t < 4; ++t) {
        int nn = n0 + 32 * t;
        uint2 o;
        o.x = pack_bf2(Ot[dg + 0][nn], Ot[dg + 1][nn]);
        o.y = pack_bf2(Ot[dg + 2][nn], Ot[dg + 3][nn]);
        *(uint2*)&OutAb[((size_t)(b * N_ + nBase + nn)) * C_ + h * D_ + dg] = o;
    }
}

// ---------------------------------------------------------------------------
// Kernels 6/7: depthwise 3x3 -> exact GELU -> depthwise 3x3, added to d_out.
// ---------------------------------------------------------------------------
__global__ __launch_bounds__(256)
void pe1_kernel(const float* __restrict__ V, const float* __restrict__ w1,
                float* __restrict__ T) {
    size_t idx = (size_t)blockIdx.x * 256 + threadIdx.x;
    int c = (int)(idx & 255);
    size_t p = idx >> 8;
    int x = (int)(p & 127); p >>= 7;
    int y = (int)(p & 127);
    int b = (int)(p >> 7);
    const float* wc = w1 + c * 9;
    float s = 0.f;
#pragma unroll
    for (int dy = -1; dy <= 1; ++dy)
#pragma unroll
        for (int dx = -1; dx <= 1; ++dx) {
            int yy = y + dy, xx = x + dx;
            if ((unsigned)yy < 128u && (unsigned)xx < 128u)
                s += wc[(dy + 1) * 3 + (dx + 1)] *
                     V[((size_t)b * N_ + yy * 128 + xx) * C_ + c];
        }
    T[idx] = 0.5f * s * (1.0f + erff(s * 0.70710678118654752f));
}

__global__ __launch_bounds__(256)
void pe2_kernel(const float* __restrict__ T, const float* __restrict__ w2,
                float* __restrict__ Out) {
    size_t idx = (size_t)blockIdx.x * 256 + threadIdx.x;
    int c = (int)(idx & 255);
    size_t p = idx >> 8;
    int x = (int)(p & 127); p >>= 7;
    int y = (int)(p & 127);
    int b = (int)(p >> 7);
    const float* wc = w2 + c * 9;
    float s = 0.f;
#pragma unroll
    for (int dy = -1; dy <= 1; ++dy)
#pragma unroll
        for (int dx = -1; dx <= 1; ++dx) {
            int yy = y + dy, xx = x + dx;
            if ((unsigned)yy < 128u && (unsigned)xx < 128u)
                s += wc[(dy + 1) * 3 + (dx + 1)] *
                     T[((size_t)b * N_ + yy * 128 + xx) * C_ + c];
        }
    Out[idx] += s;
}

// ---------------------------------------------------------------------------
extern "C" void kernel_launch(void* const* d_in, const int* in_sizes, int n_in,
                              void* d_out, int out_size, void* d_ws, size_t ws_size,
                              hipStream_t stream) {
    const float* x     = (const float*)d_in[0];
    const float* y     = (const float*)d_in[1];
    const float* Wq    = (const float*)d_in[2];
    const float* Wk    = (const float*)d_in[3];
    const float* Wv    = (const float*)d_in[4];
    const float* Wqy   = (const float*)d_in[5];
    const float* Wky   = (const float*)d_in[6];
    const float* resc  = (const float*)d_in[7];
    const float* rescy = (const float*)d_in[8];
    const float* projw = (const float*)d_in[9];
    const float* projb = (const float*)d_in[10];
    const float* pew1  = (const float*)d_in[11];
    const float* pew2  = (const float*)d_in[12];
    float* out = (float*)d_out;

    char* w = (char*)d_ws;
    const size_t SZP = (size_t)B_ * N_ * C_ * sizeof(float);     // 64 MiB
    const size_t SZH = SZP / 2;                                  // bf16 tensor
    const size_t SZW = (size_t)C_ * C_ * sizeof(unsigned short); // 128 KiB

    float* Pq  = (float*)(w + 0 * SZP);
    float* Pk  = (float*)(w + 1 * SZP);
    float* Pv  = (float*)(w + 2 * SZP);
    float* Pqy = (float*)(w + 3 * SZP);
    float* Pky = (float*)(w + 4 * SZP);
    char* cur = w + 5 * SZP;
    unsigned short* Qn   = (unsigned short*)cur; cur += SZH;
    unsigned short* Kn   = (unsigned short*)cur; cur += SZH;
    unsigned short* Qyn  = (unsigned short*)cur; cur += SZH;
    unsigned short* Kyn  = (unsigned short*)cur; cur += SZH;
    unsigned short* Xb   = (unsigned short*)cur; cur += SZH;
    unsigned short* Yb   = (unsigned short*)cur; cur += SZH;
    unsigned short* OutAb= (unsigned short*)cur; cur += SZH;
    float* Gt            = (float*)cur;          cur += SZP;
    float* attnX         = (float*)cur;          cur += (1 << 17);
    float* attnY         = (float*)cur;          cur += (1 << 17);
    unsigned short* xo2b = (unsigned short*)cur; cur += (1 << 16);
    unsigned short* Wqb  = (unsigned short*)cur; cur += SZW;
    unsigned short* Wkb  = (unsigned short*)cur; cur += SZW;
    unsigned short* Wvb  = (unsigned short*)cur; cur += SZW;
    unsigned short* Wqyb = (unsigned short*)cur; cur += SZW;
    unsigned short* Wkyb = (unsigned short*)cur; cur += SZW;
    unsigned short* Wpb  = (unsigned short*)cur; cur += SZW;

    const int M  = B_ * N_;                 // 65536
    const int n4Big = (B_ * N_ * C_) / 4;
    const int n4W   = (C_ * C_) / 4;

    // 0) one-time fp32 -> bf16 conversions
    cvt_bf16_kernel<<<n4Big / 256, 256, 0, stream>>>(x, Xb, n4Big);
    cvt_bf16_kernel<<<n4Big / 256, 256, 0, stream>>>(y, Yb, n4Big);
    cvt_bf16_kernel<<<(n4W + 255) / 256, 256, 0, stream>>>(Wq,    Wqb,  n4W);
    cvt_bf16_kernel<<<(n4W + 255) / 256, 256, 0, stream>>>(Wk,    Wkb,  n4W);
    cvt_bf16_kernel<<<(n4W + 255) / 256, 256, 0, stream>>>(Wv,    Wvb,  n4W);
    cvt_bf16_kernel<<<(n4W + 255) / 256, 256, 0, stream>>>(Wqy,   Wqyb, n4W);
    cvt_bf16_kernel<<<(n4W + 255) / 256, 256, 0, stream>>>(Wky,   Wkyb, n4W);
    cvt_bf16_kernel<<<(n4W + 255) / 256, 256, 0, stream>>>(projw, Wpb,  n4W);

    dim3 gG(M / 128, C_ / 128), bG(256);

    // 1) five projections (bf16 WMMA, f32 accumulate)
    gemm256_kernel<<<gG, bG, 0, stream>>>(Xb, Wqb,  nullptr, Pq,  M);
    gemm256_kernel<<<gG, bG, 0, stream>>>(Xb, Wkb,  nullptr, Pk,  M);
    gemm256_kernel<<<gG, bG, 0, stream>>>(Xb, Wvb,  nullptr, Pv,  M);
    gemm256_kernel<<<gG, bG, 0, stream>>>(Yb, Wqyb, nullptr, Pqy, M);
    gemm256_kernel<<<gG, bG, 0, stream>>>(Yb, Wkyb, nullptr, Pky, M);

    // 2) normalize along n, transpose to [b*c, n] bf16
    l2norm_kernel<<<B_ * C_, 256, 0, stream>>>(Pq,  Qn);
    l2norm_kernel<<<B_ * C_, 256, 0, stream>>>(Pk,  Kn);
    l2norm_kernel<<<B_ * C_, 256, 0, stream>>>(Pqy, Qyn);
    l2norm_kernel<<<B_ * C_, 256, 0, stream>>>(Pky, Kyn);

    // 3) channel attention matrices
    qk_softmax_kernel<<<B_ * H_, 256, 0, stream>>>(Kn,  Qn,  resc,  attnX);
    qk_softmax_kernel<<<B_ * H_, 256, 0, stream>>>(Kyn, Qyn, rescy, attnY);

    // 4) xo = attny @ attn
    xo_kernel<<<B_ * H_, 32, 0, stream>>>(attnY, attnX, xo2b);

    // 5) xo @ v -> [b,n,c] bf16
    av_kernel<<<dim3(N_ / 128, B_ * H_), 256, 0, stream>>>(xo2b, Pv, OutAb);

    // 6) output projection + bias -> d_out
    gemm256_kernel<<<gG, bG, 0, stream>>>(OutAb, Wpb, projb, out, M);

    // 7) positional branch added into d_out
    const int nElem = B_ * N_ * C_;
    pe1_kernel<<<nElem / 256, 256, 0, stream>>>(Pv, pew1, Gt);
    pe2_kernel<<<nElem / 256, 256, 0, stream>>>(Gt, pew2, out);
}
